// FABlock2D_20220706030123
// MI455X (gfx1250) — matrix operand order repaired and checked by reading, not compiled, not run on hardware
//
#include <hip/hip_runtime.h>
#include <hip/hip_bf16.h>
#include <math.h>

// ---------------------------------------------------------------------------
// FABlock2D for MI455X (gfx1250, wave32).
// Heavy matmuls on v_wmma_f32_16x16x32_f16 (f16 in, f32 accumulate).
// kx-apply and ky-apply are FUSED per (b,head,c) plane through 320KB LDS:
//   T = kx @ U kept entirely in LDS (128KB), O = T @ ky^T streamed out as f16,
//   instance-norm stats folded into the same workgroup (it owns the plane).
// U plane staged global->LDS via the Tensor Data Mover when available.
// ---------------------------------------------------------------------------

#define EPSF 1e-5f

typedef __attribute__((ext_vector_type(16))) _Float16 v16h;
typedef __attribute__((ext_vector_type(8)))  float    v8f;
typedef __attribute__((ext_vector_type(4)))  unsigned int u32x4;
typedef __attribute__((ext_vector_type(8)))  int      i32x8;
typedef __attribute__((ext_vector_type(4)))  int      i32x4;

__device__ __forceinline__ v8f wmma_f16(v16h a, v16h b, v8f c) {
  return __builtin_amdgcn_wmma_f32_16x16x32_f16(false, a, false, b, (short)0, c,
                                                false, false);
}

__device__ __forceinline__ float gelu_erf(float x) {
  return 0.5f * x * (1.0f + erff(x * 0.70710678118654752f));
}

// A fragment (16x32, MxK), row-major source; lane = M row.
// lanes 0-15: halves 0..15 -> K 0..15
// lanes 16-31: halves 0..15 -> K 16..31
__device__ __forceinline__ v16h frag_a_rm(const _Float16* A, int lda, int row,
                                          int k0, int lane) {
  const _Float16* p = A + (size_t)row * lda + k0;
  int hi = (lane >> 4) & 1;
  int b0 = hi ? 16 : 0, b1 = hi ? 24 : 8;
  v16h r;
#pragma unroll
  for (int h = 0; h < 8; ++h) { r[h] = p[b0 + h]; r[8 + h] = p[b1 + h]; }
  return r;
}

// B fragment (32x16, KxN) from row-major [N][K] source (B = srcT). Contiguous K.
__device__ __forceinline__ v16h frag_b_nk(const _Float16* Bm, int ldb, int n0,
                                          int k0, int lane) {
  const _Float16* p = Bm + (size_t)(n0 + (lane & 15)) * ldb + k0 +
                      (((lane >> 4) & 1) << 4);
  v16h r;
#pragma unroll
  for (int h = 0; h < 16; ++h) r[h] = p[h];
  return r;
}

// B fragment from row-major [K][N] source (K strided).
__device__ __forceinline__ v16h frag_b_kn(const _Float16* Bm, size_t ldb, int n0,
                                          int k0, int lane) {
  const _Float16* p = Bm + (size_t)(k0 + (((lane >> 4) & 1) << 4)) * ldb + n0 +
                      (lane & 15);
  v16h r;
#pragma unroll
  for (int h = 0; h < 16; ++h) r[h] = p[(size_t)h * ldb];
  return r;
}

// D (16x16 f32): VGPR r -> (M=mb+r, N=lane&15), mb = 8*(lane>=16).
__device__ __forceinline__ void store_d_f16(_Float16* C, size_t ldc, int r0,
                                            int c0, const v8f& d, int lane) {
  int n = lane & 15, mb = ((lane >> 4) & 1) * 8;
#pragma unroll
  for (int r = 0; r < 8; ++r)
    C[(size_t)(r0 + mb + r) * ldc + c0 + n] = (_Float16)d[r];
}

// ---------------------------------------------------------------------------
// 1. GroupNorm statistics (two stage)
// ---------------------------------------------------------------------------
__global__ void k_gn_part(const float* __restrict__ u, float* __restrict__ part) {
  __shared__ float r1[256], r2[256];
  int blk = blockIdx.x;  // b*64 + c plane
  int t = threadIdx.x;
  const float* p = u + ((size_t)blk << 16);
  float s = 0.f, s2 = 0.f;
  for (int i = 0; i < 256; ++i) {
    float v = p[(size_t)i * 256 + t];
    s += v; s2 += v * v;
  }
  r1[t] = s; r2[t] = s2; __syncthreads();
  for (int w = 128; w > 0; w >>= 1) {
    if (t < w) { r1[t] += r1[t + w]; r2[t] += r2[t + w]; }
    __syncthreads();
  }
  if (t == 0) { part[2 * blk] = r1[0]; part[2 * blk + 1] = r2[0]; }
}

__global__ void k_gn_final(const float* __restrict__ part,
                           float* __restrict__ gstats) {
  __shared__ float r1[64], r2[64];
  int b = blockIdx.x, t = threadIdx.x;
  r1[t] = part[2 * (b * 64 + t)];
  r2[t] = part[2 * (b * 64 + t) + 1];
  __syncthreads();
  for (int w = 32; w > 0; w >>= 1) {
    if (t < w) { r1[t] += r1[t + w]; r2[t] += r2[t + w]; }
    __syncthreads();
  }
  if (t == 0) {
    float n = 64.f * 65536.f;
    float m = r1[0] / n;
    float v = r2[0] / n - m * m;
    gstats[2 * b] = m;
    gstats[2 * b + 1] = rsqrtf(v + EPSF);
  }
}

// ---------------------------------------------------------------------------
// 2. Fused GroupNorm + input projections (WMMA). M = 192 out channels
//    (0..127 -> u_phi f16, 128..191 -> ui f32), N = pixels, K = 64.
// ---------------------------------------------------------------------------
__global__ void k_proj(const float* __restrict__ u, const float* __restrict__ gnw,
                       const float* __restrict__ gnb, const float* __restrict__ win,
                       const float* __restrict__ wto, const float* __restrict__ gstats,
                       _Float16* __restrict__ uphi, float* __restrict__ ui) {
  int wid = blockIdx.x * 8 + (threadIdx.x >> 5);
  int lane = threadIdx.x & 31;
  int mt = wid % 12, nt = wid / 12;
  size_t p0 = (size_t)nt * 16;
  int b = (int)(p0 >> 16), off0 = (int)(p0 & 65535);
  float mean = gstats[2 * b], inv = gstats[2 * b + 1];
  int hi = (lane >> 4) & 1, ln = lane & 15;
  int orow = mt * 16 + ln;
  const float* Wr = (orow < 128) ? (win + orow * 64) : (wto + (orow - 128) * 64);
  int n_ = off0 + ln;
  v8f acc = {};
#pragma unroll
  for (int ks = 0; ks < 2; ++ks) {
    int k0 = ks * 32;
    v16h a, bb;
    int a0 = k0 + (hi ? 16 : 0), a1 = k0 + (hi ? 24 : 8);
#pragma unroll
    for (int h = 0; h < 8; ++h) {
      a[h] = (_Float16)Wr[a0 + h];
      a[8 + h] = (_Float16)Wr[a1 + h];
    }
    int cb = k0 + hi * 16;
#pragma unroll
    for (int h = 0; h < 16; ++h) {
      int c = cb + h;
      float v = (u[((size_t)(b * 64 + c) << 16) + n_] - mean) * inv * gnw[c] + gnb[c];
      bb[h] = (_Float16)v;
    }
    acc = wmma_f16(a, bb, acc);
  }
  int mb = hi * 8;
  if (mt < 8) {
#pragma unroll
    for (int r = 0; r < 8; ++r) {
      int o = mt * 16 + mb + r;
      uphi[((size_t)(b * 128 + o) << 16) + off0 + ln] = (_Float16)acc[r];
    }
  } else {
#pragma unroll
    for (int r = 0; r < 8; ++r) {
      int o = mt * 16 + mb + r - 128;
      ui[((size_t)(b * 64 + o) << 16) + off0 + ln] = acc[r];
    }
  }
}

// ---------------------------------------------------------------------------
// 3. Axis mean pools of ui
// ---------------------------------------------------------------------------
__global__ void k_pool_x(const float* __restrict__ ui, float* __restrict__ px) {
  int b = blockIdx.x >> 8, x = blockIdx.x & 255, c = threadIdx.x;
  const float* p = ui + ((size_t)(b * 64 + c) << 16) + (size_t)x * 256;
  float s = 0.f;
  for (int y = 0; y < 256; ++y) s += p[y];
  px[((size_t)(b * 256 + x)) * 64 + c] = s * (1.f / 256.f);
}
__global__ void k_pool_y(const float* __restrict__ ui, float* __restrict__ py) {
  int b = blockIdx.x >> 8, y = blockIdx.x & 255, c = threadIdx.x;
  const float* p = ui + ((size_t)(b * 64 + c) << 16) + y;
  float s = 0.f;
  for (int x = 0; x < 256; ++x) s += p[(size_t)x * 256];
  py[((size_t)(b * 256 + y)) * 64 + c] = s * (1.f / 256.f);
}

// ---------------------------------------------------------------------------
// 4. Pooling reducer MLP: Linear(64) -> LN -> FFN(64->128 gelu ->128)+b
// ---------------------------------------------------------------------------
__global__ void k_reducer(const float* __restrict__ poolx, const float* __restrict__ pooly,
                          const float* __restrict__ xwin, const float* __restrict__ xlg,
                          const float* __restrict__ xlb, const float* __restrict__ xw1,
                          const float* __restrict__ xw2, const float* __restrict__ xb2,
                          const float* __restrict__ ywin, const float* __restrict__ ylg,
                          const float* __restrict__ ylb, const float* __restrict__ yw1,
                          const float* __restrict__ yw2, const float* __restrict__ yb2,
                          float* __restrict__ ulat) {
  __shared__ float pm[64], hh[64], f1[128], mv[2];
  int blk = blockIdx.x;
  int s = blk >> 10, rem = blk & 1023;
  int b = rem >> 8, n = rem & 255;
  int t = threadIdx.x;
  const float* pool = s ? pooly : poolx;
  const float* win = s ? ywin : xwin;
  const float* lg = s ? ylg : xlg;
  const float* lb = s ? ylb : xlb;
  const float* w1 = s ? yw1 : xw1;
  const float* w2 = s ? yw2 : xw2;
  const float* b2 = s ? yb2 : xb2;
  if (t < 64) pm[t] = pool[((size_t)(b * 256 + n)) * 64 + t];
  __syncthreads();
  if (t < 64) {
    float acc = 0.f;
    const float* wr = win + t * 64;
    for (int c = 0; c < 64; ++c) acc += pm[c] * wr[c];
    hh[t] = acc;
  }
  __syncthreads();
  if (t == 0) {
    float m = 0.f, v = 0.f;
    for (int j = 0; j < 64; ++j) m += hh[j];
    m *= (1.f / 64.f);
    for (int j = 0; j < 64; ++j) { float d = hh[j] - m; v += d * d; }
    v *= (1.f / 64.f);
    mv[0] = m; mv[1] = rsqrtf(v + EPSF);
  }
  __syncthreads();
  if (t < 64) hh[t] = (hh[t] - mv[0]) * mv[1] * lg[t] + lb[t];
  __syncthreads();
  {
    float acc = 0.f;
    const float* wr = w1 + t * 64;
    for (int j = 0; j < 64; ++j) acc += hh[j] * wr[j];
    f1[t] = gelu_erf(acc);
  }
  __syncthreads();
  {
    float acc = b2[t];
    const float* wr = w2 + t * 128;
    for (int k = 0; k < 128; ++k) acc += f1[k] * wr[k];
    ulat[(((size_t)s * 4 + b) * 256 + n) * 128 + t] = acc;
  }
}

// ---------------------------------------------------------------------------
// 5. qk projection + RoPE -> q,k (f16) per head
// ---------------------------------------------------------------------------
__global__ void k_qkrope(const float* __restrict__ ulat, const float* __restrict__ kxw,
                         const float* __restrict__ kyw, _Float16* __restrict__ qh,
                         _Float16* __restrict__ kh) {
  __shared__ float lat[128], qk[512];
  int blk = blockIdx.x;
  int s = blk >> 10, rem = blk & 1023;
  int b = rem >> 8, n = rem & 255;
  int t = threadIdx.x;
  const float* wqk = s ? kyw : kxw;
  lat[t] = ulat[(((size_t)s * 4 + b) * 256 + n) * 128 + t];
  __syncthreads();
#pragma unroll
  for (int part = 0; part < 4; ++part) {
    int o = t + part * 128;
    float acc = 0.f;
    const float* wr = wqk + (size_t)o * 128;
    for (int c = 0; c < 128; ++c) acc += lat[c] * wr[c];
    qk[o] = acc;
  }
  __syncthreads();
  float pos = ((float)n / 255.f) * 64.f;  // linspace(0,1)/min_freq
#pragma unroll
  for (int part = 0; part < 2; ++part) {
    int idx = t + part * 128;  // (head, d)
    int h = idx >> 6, d = idx & 63, j = d & 31;
    float f = pos * powf(10000.f, -((float)(2 * j)) / 64.f);
    float cf = cosf(f), sf = sinf(f);
    float qv = qk[h * 64 + d];
    float qr = (d < 32) ? -qk[h * 64 + d + 32] : qk[h * 64 + d - 32];
    float kv = qk[256 + h * 64 + d];
    float kr = (d < 32) ? -qk[256 + h * 64 + d + 32] : qk[256 + h * 64 + d - 32];
    size_t base = (((size_t)(s * 16 + b * 4 + h)) * 256 + n) * 64 + d;
    qh[base] = (_Float16)(qv * cf + qr * sf);
    kh[base] = (_Float16)(kv * cf + kr * sf);
  }
}

// ---------------------------------------------------------------------------
// 6. kx/ky = q @ k^T per (set,b,head) — WMMA, K=64
// ---------------------------------------------------------------------------
__global__ void k_kxky(const _Float16* __restrict__ qh, const _Float16* __restrict__ kh,
                       _Float16* __restrict__ kxh, _Float16* __restrict__ kyh) {
  int wid = blockIdx.x * 8 + (threadIdx.x >> 5);
  int lane = threadIdx.x & 31;
  int jt = wid & 15, it = (wid >> 4) & 15, sb = wid >> 8;  // s*16+bh
  int s = sb >> 4, bh = sb & 15;
  const _Float16* A = qh + ((size_t)sb) * 256 * 64;
  const _Float16* Bm = kh + ((size_t)sb) * 256 * 64;
  _Float16* C = (s ? kyh : kxh) + ((size_t)bh << 16);
  int row = it * 16 + (lane & 15);
  v8f acc = {};
#pragma unroll
  for (int ks = 0; ks < 2; ++ks) {
    v16h a = frag_a_rm(A, 64, row, ks * 32, lane);
    v16h bb = frag_b_nk(Bm, 64, jt * 16, ks * 32, lane);
    acc = wmma_f16(a, bb, acc);
  }
  store_d_f16(C, 256, it * 16, jt * 16, acc, lane);
}

// ---------------------------------------------------------------------------
// 7. FUSED plane kernel: per pl=(b,ch):
//      stage U plane (128KB) -> LDS (TDM when available)
//      T = kx @ U           -> LDS (128KB, never touches HBM)
//      O = T @ ky^T         -> global f16, with plane mean/rstd reduced here
//    8 waves, each owns 2 row-strips of 16 rows; A-fragments hoisted and
//    reused across all 16 column tiles (256 WMMAs per wave per phase).
// ---------------------------------------------------------------------------
__global__ void k_apply(const _Float16* __restrict__ kxh, const _Float16* __restrict__ kyh,
                        const _Float16* __restrict__ uphi, _Float16* __restrict__ O,
                        float* __restrict__ instat) {
  extern __shared__ char smem[];
  _Float16* Ulds = (_Float16*)smem;             // 256*256 f16 = 131072 B
  _Float16* Tlds = (_Float16*)(smem + 131072);  // 256*256 f16 = 131072 B
  float* red1 = (float*)(smem + 262144);        // 256 f32
  float* red2 = red1 + 256;                     // 256 f32

  int pl = blockIdx.x;
  int b = pl >> 7, ch = pl & 127, h = ch >> 5;
  int wv = threadIdx.x >> 5, lane = threadIdx.x & 31;
  int ln = lane & 15, mb = ((lane >> 4) & 1) * 8;

  const _Float16* Usrc = uphi + ((size_t)pl << 16);

  // ---- stage U plane into LDS --------------------------------------------
#if defined(__gfx1250__) && __has_builtin(__builtin_amdgcn_tensor_load_to_lds) && \
    __has_builtin(__builtin_amdgcn_s_wait_tensorcnt)
  if (wv == 0) {
    unsigned long long ga = (unsigned long long)(uintptr_t)Usrc;
    unsigned int ldsoff = (unsigned int)(uintptr_t)Ulds;  // low 32 bits = LDS offset
    u32x4 g0;
    g0[0] = 1u;                                   // count=1, user descriptor
    g0[1] = ldsoff;                               // lds_addr
    g0[2] = (unsigned int)(ga & 0xffffffffu);     // global_addr[31:0]
    g0[3] = (unsigned int)((ga >> 32) & 0x1ffffffu) | (2u << 30);  // addr[56:32]|type=2
    i32x8 g1;
    g1[0] = (int)(1u << 16);        // workgroup_mask=0, data_size=2B
    g1[1] = (int)(256u << 16);      // tensor_dim0[15:0]=256 (bits 63:48)
    g1[2] = (int)(256u << 16);      // tensor_dim0[31:16]=0, tensor_dim1[15:0]=256
    g1[3] = (int)(256u << 16);      // tensor_dim1[31:16]=0, tile_dim0=256
    g1[4] = (int)256u;              // tile_dim1=256, tile_dim2=0
    g1[5] = (int)256u;              // tensor_dim0_stride[31:0]=256
    g1[6] = 0;                      // stride0[47:32]=0, stride1[15:0]=0
    g1[7] = 1;                      // stride1[47:16]=1 -> stride1=65536
    i32x4 gz4 = {};
    i32x8 gz8 = {};
    __builtin_amdgcn_tensor_load_to_lds(g0, g1, gz4, gz4, gz8, 0);
    __builtin_amdgcn_s_wait_tensorcnt(0);
  }
#else
  {
    const uint4* src = (const uint4*)Usrc;
    uint4* dst = (uint4*)Ulds;
    for (int i = threadIdx.x; i < 8192; i += 256) dst[i] = src[i];
  }
#endif
  __syncthreads();

  // ---- phase 1: T = kx @ U (into LDS) ------------------------------------
  const _Float16* A1 = kxh + ((size_t)(b * 4 + h) << 16);
#pragma unroll
  for (int i2 = 0; i2 < 2; ++i2) {
    int it = wv * 2 + i2;
    int row = it * 16 + ln;
    v16h afr[8];
#pragma unroll
    for (int ks = 0; ks < 8; ++ks) afr[ks] = frag_a_rm(A1, 256, row, ks * 32, lane);
    for (int mt = 0; mt < 16; ++mt) {
      v8f acc = {};
#pragma unroll
      for (int ks = 0; ks < 8; ++ks)
        acc = wmma_f16(afr[ks], frag_b_kn(Ulds, 256, mt * 16, ks * 32, lane), acc);
#pragma unroll
      for (int r = 0; r < 8; ++r)
        Tlds[(size_t)(it * 16 + mb + r) * 256 + mt * 16 + ln] = (_Float16)acc[r];
    }
  }
  __syncthreads();

  // ---- phase 2: O = T @ ky^T (+ instance-norm stats) ---------------------
  const _Float16* B2 = kyh + ((size_t)(b * 4 + h) << 16);
  _Float16* Op = O + ((size_t)pl << 16);
  float s = 0.f, s2 = 0.f;
#pragma unroll
  for (int i2 = 0; i2 < 2; ++i2) {
    int it = wv * 2 + i2;
    int row = it * 16 + ln;
    v16h afr[8];
#pragma unroll
    for (int ks = 0; ks < 8; ++ks) afr[ks] = frag_a_rm(Tlds, 256, row, ks * 32, lane);
    for (int lt = 0; lt < 16; ++lt) {
      v8f acc = {};
#pragma unroll
      for (int ks = 0; ks < 8; ++ks)
        acc = wmma_f16(afr[ks], frag_b_nk(B2, 256, lt * 16, ks * 32, lane), acc);
#pragma unroll
      for (int r = 0; r < 8; ++r) {
        float v = acc[r];
        s += v; s2 += v * v;
        Op[(size_t)(it * 16 + mb + r) * 256 + lt * 16 + ln] = (_Float16)v;
      }
    }
  }
  // plane-level stats reduction (workgroup owns the whole plane)
  red1[threadIdx.x] = s; red2[threadIdx.x] = s2;
  __syncthreads();
  for (int w = 128; w > 0; w >>= 1) {
    if ((int)threadIdx.x < w) {
      red1[threadIdx.x] += red1[threadIdx.x + w];
      red2[threadIdx.x] += red2[threadIdx.x + w];
    }
    __syncthreads();
  }
  if (threadIdx.x == 0) {
    float m = red1[0] * (1.f / 65536.f);
    float v = red2[0] * (1.f / 65536.f) - m * m;
    instat[2 * pl] = m;
    instat[2 * pl + 1] = rsqrtf(v + EPSF);
  }
}

// ---------------------------------------------------------------------------
// 8. FUSED output convs: norm(O) -> conv1(128->64) -> GELU -> conv2(64->64)
//    per 64-pixel strip through an 8KB LDS tile; + residual u.
// ---------------------------------------------------------------------------
__global__ void k_outc(const _Float16* __restrict__ O, const float* __restrict__ st,
                       const float* __restrict__ w1, const float* __restrict__ w2,
                       const float* __restrict__ u, float* __restrict__ out) {
  __shared__ _Float16 G[64 * 64];  // [ch][pix]
  int wv = threadIdx.x >> 5, lane = threadIdx.x & 31;
  int ln = lane & 15, hi = (lane >> 4) & 1, mb = hi * 8;
  size_t p0 = (size_t)blockIdx.x * 64;
  int b = (int)(p0 >> 16), off0 = (int)(p0 & 65535);

  // phase A: G = gelu(w1 @ instnorm(O))
#pragma unroll
  for (int q = 0; q < 2; ++q) {
    int tid2 = wv * 2 + q;
    int mt = tid2 & 3, nt4 = tid2 >> 2;
    int ncol = off0 + nt4 * 16 + ln;
    const float* Wr = w1 + (mt * 16 + ln) * 128;
    v8f acc = {};
#pragma unroll
    for (int ks = 0; ks < 4; ++ks) {
      int k0 = ks * 32;
      v16h a, bb;
      int a0 = k0 + (hi ? 16 : 0), a1 = k0 + (hi ? 24 : 8);
#pragma unroll
      for (int hh = 0; hh < 8; ++hh) {
        a[hh] = (_Float16)Wr[a0 + hh];
        a[8 + hh] = (_Float16)Wr[a1 + hh];
      }
      int cb = k0 + hi * 16;
#pragma unroll
      for (int hh = 0; hh < 16; ++hh) {
        int pli = b * 128 + cb + hh;
        float v = ((float)O[((size_t)pli << 16) + ncol] - st[2 * pli]) * st[2 * pli + 1];
        bb[hh] = (_Float16)v;
      }
      acc = wmma_f16(a, bb, acc);
    }
#pragma unroll
    for (int r = 0; r < 8; ++r)
      G[(mt * 16 + mb + r) * 64 + nt4 * 16 + ln] = (_Float16)gelu_erf(acc[r]);
  }
  __syncthreads();

  // phase B: out = w2 @ G + u
#pragma unroll
  for (int q = 0; q < 2; ++q) {
    int tid2 = wv * 2 + q;
    int mt = tid2 & 3, nt4 = tid2 >> 2;
    const float* Wr = w2 + (mt * 16 + ln) * 64;
    v8f acc = {};
#pragma unroll
    for (int ks = 0; ks < 2; ++ks) {
      int k0 = ks * 32;
      v16h a;
      int a0 = k0 + (hi ? 16 : 0), a1 = k0 + (hi ? 24 : 8);
#pragma unroll
      for (int hh = 0; hh < 8; ++hh) {
        a[hh] = (_Float16)Wr[a0 + hh];
        a[8 + hh] = (_Float16)Wr[a1 + hh];
      }
      v16h bb = frag_b_kn(G, (size_t)64, nt4 * 16, k0, lane);
      acc = wmma_f16(a, bb, acc);
    }
#pragma unroll
    for (int r = 0; r < 8; ++r) {
      int o = mt * 16 + mb + r;
      size_t idx = ((size_t)(b * 64 + o) << 16) + off0 + nt4 * 16 + ln;
      out[idx] = acc[r] + u[idx];
    }
  }
}

// ---------------------------------------------------------------------------
// Workspace layout (bytes)
// ---------------------------------------------------------------------------
static constexpr size_t OFF_UPHI   = 0;                                    // f16 [4,128,256,256]
static constexpr size_t OFF_UI     = OFF_UPHI   + (size_t)4*128*65536*2;   // f32 [4,64,256,256]
static constexpr size_t OFF_O      = OFF_UI     + (size_t)4*64*65536*4;    // f16 [4,128,256,256]
static constexpr size_t OFF_QH     = OFF_O      + (size_t)4*128*65536*2;   // f16 [2,4,4,256,64]
static constexpr size_t OFF_KH     = OFF_QH     + (size_t)2*16*256*64*2;
static constexpr size_t OFF_KXH    = OFF_KH     + (size_t)2*16*256*64*2;   // f16 [16,256,256]
static constexpr size_t OFF_KYH    = OFF_KXH    + (size_t)16*65536*2;
static constexpr size_t OFF_POOLX  = OFF_KYH    + (size_t)16*65536*2;      // f32 [4,256,64]
static constexpr size_t OFF_POOLY  = OFF_POOLX  + (size_t)4*256*64*4;
static constexpr size_t OFF_ULAT   = OFF_POOLY  + (size_t)4*256*64*4;      // f32 [2,4,256,128]
static constexpr size_t OFF_GNPART = OFF_ULAT   + (size_t)2*4*256*128*4;
static constexpr size_t OFF_GSTATS = OFF_GNPART + (size_t)256*2*4;
static constexpr size_t OFF_INST   = OFF_GSTATS + (size_t)8*4;

extern "C" void kernel_launch(void* const* d_in, const int* in_sizes, int n_in,
                              void* d_out, int out_size, void* d_ws, size_t ws_size,
                              hipStream_t stream) {
  const float* u        = (const float*)d_in[0];
  const float* gn_w     = (const float*)d_in[1];
  const float* gn_b     = (const float*)d_in[2];
  const float* w_inproj = (const float*)d_in[3];
  const float* w_to_in  = (const float*)d_in[4];
  const float* px_win   = (const float*)d_in[5];
  const float* px_ln_g  = (const float*)d_in[6];
  const float* px_ln_b  = (const float*)d_in[7];
  const float* px_w1    = (const float*)d_in[8];
  const float* px_w2    = (const float*)d_in[9];
  const float* px_b2    = (const float*)d_in[10];
  const float* py_win   = (const float*)d_in[11];
  const float* py_ln_g  = (const float*)d_in[12];
  const float* py_ln_b  = (const float*)d_in[13];
  const float* py_w1    = (const float*)d_in[14];
  const float* py_w2    = (const float*)d_in[15];
  const float* py_b2    = (const float*)d_in[16];
  const float* kx_wqk   = (const float*)d_in[17];
  const float* ky_wqk   = (const float*)d_in[18];
  const float* out_w1   = (const float*)d_in[19];
  const float* out_w2   = (const float*)d_in[20];
  float* out = (float*)d_out;

  char* ws = (char*)d_ws;
  _Float16*  uphi   = (_Float16*)(ws + OFF_UPHI);
  float*     ui     = (float*)(ws + OFF_UI);
  _Float16*  Obuf   = (_Float16*)(ws + OFF_O);
  _Float16*  qh     = (_Float16*)(ws + OFF_QH);
  _Float16*  kh     = (_Float16*)(ws + OFF_KH);
  _Float16*  kxh    = (_Float16*)(ws + OFF_KXH);
  _Float16*  kyh    = (_Float16*)(ws + OFF_KYH);
  float*     poolx  = (float*)(ws + OFF_POOLX);
  float*     pooly  = (float*)(ws + OFF_POOLY);
  float*     ulat   = (float*)(ws + OFF_ULAT);
  float*     gnpart = (float*)(ws + OFF_GNPART);
  float*     gstats = (float*)(ws + OFF_GSTATS);
  float*     instat = (float*)(ws + OFF_INST);

  k_gn_part <<<256, 256, 0, stream>>>(u, gnpart);
  k_gn_final<<<4, 64, 0, stream>>>(gnpart, gstats);
  k_proj    <<<24576, 256, 0, stream>>>(u, gn_w, gn_b, w_inproj, w_to_in, gstats,
                                        uphi, ui);
  k_pool_x  <<<1024, 64, 0, stream>>>(ui, poolx);
  k_pool_y  <<<1024, 64, 0, stream>>>(ui, pooly);
  k_reducer <<<2048, 128, 0, stream>>>(poolx, pooly,
                                       px_win, px_ln_g, px_ln_b, px_w1, px_w2, px_b2,
                                       py_win, py_ln_g, py_ln_b, py_w1, py_w2, py_b2,
                                       ulat);
  k_qkrope  <<<2048, 128, 0, stream>>>(ulat, kx_wqk, ky_wqk, qh, kh);
  k_kxky    <<<1024, 256, 0, stream>>>(qh, kh, kxh, kyh);
  // 131072 (U) + 131072 (T) + 2048 (reduce) bytes of LDS per workgroup
  k_apply   <<<512, 256, 264192, stream>>>(kxh, kyh, uphi, Obuf, instat);
  k_outc    <<<4096, 256, 0, stream>>>(Obuf, instat, out_w1, out_w2, u, out);
}
